// BaichuanAttention_4801773437527
// MI455X (gfx1250) — matrix operand order repaired and checked by reading, not compiled, not run on hardware
//
#include <hip/hip_runtime.h>

// ---------------------------------------------------------------------------
// Problem constants (from reference)
// ---------------------------------------------------------------------------
constexpr int B_   = 2;
constexpr int S_   = 2048;
constexpr int HID_ = 4096;
constexpr int H_   = 32;
constexpr int KV_  = 8;
constexpr int D_   = 128;
constexpr int F_   = (H_ + 2 * KV_) * D_;   // 6144
constexpr int WIN_ = 512;
constexpr int M_   = B_ * S_;               // 4096 rows

typedef __attribute__((ext_vector_type(4)))  __bf16 bf16x4;
typedef __attribute__((ext_vector_type(8)))  __bf16 bf16x8;
typedef __attribute__((ext_vector_type(16))) __bf16 bf16x16;
typedef __attribute__((ext_vector_type(8)))  float  f32x8;

__device__ __forceinline__ bf16x16 comb16(const __bf16* p0, const __bf16* p1) {
  bf16x8 lo = *(const bf16x8*)p0;
  bf16x8 hi = *(const bf16x8*)p1;
  return __builtin_shufflevector(lo, hi, 0,1,2,3,4,5,6,7,8,9,10,11,12,13,14,15);
}

__device__ __forceinline__ f32x8 wmma_bf16(bf16x16 a, bf16x16 b, f32x8 c) {
  // D = A(16x32 bf16) * B(32x16 bf16) + C(16x16 f32)
  return __builtin_amdgcn_wmma_f32_16x16x32_bf16(false, a, false, b,
                                                 (short)0, c, false, false);
}

// Async DMA: one lane moves 16B global -> LDS, tracked by ASYNCcnt.
__device__ __forceinline__ void async_copy_b128(const void* gaddr, void* lds) {
  unsigned lds_off = (unsigned)(uintptr_t)lds;   // low 32 bits = LDS offset
  asm volatile("global_load_async_to_lds_b128 %0, %1, off"
               :: "v"(lds_off), "v"(gaddr) : "memory");
}
__device__ __forceinline__ void wait_asynccnt0() {
  asm volatile("s_wait_asynccnt 0x0" ::: "memory");
}

// ---------------------------------------------------------------------------
// Tiled bf16 WMMA GEMM:  C[M,N](f32) = A[M,K] * B[K,N](f32)
// A may be f32 (converted while staging) or bf16.
// Block = 128 threads (4 waves), block tile 128x128, K-step 32.
// Register-buffered pipeline: next tile's global loads issue during WMMA.
// ---------------------------------------------------------------------------
template <typename TA>
__global__ __launch_bounds__(128) void gemm_bf16_wmma_kernel(
    const TA* __restrict__ A, const float* __restrict__ Bm,
    float* __restrict__ C, int M, int N, int K)
{
  __shared__ __bf16 sA [128][40];   // [m][k]  (+8 pad)
  __shared__ __bf16 sBt[128][40];   // [n][k]  transposed B tile (+8 pad)

  const int tid  = threadIdx.x;
  const int lane = tid & 31;
  const int wave = tid >> 5;
  const int l16  = lane & 15;
  const int lh   = lane >> 4;          // lane half (0/1)
  const int wm   = (wave >> 1) * 64;
  const int wn   = (wave & 1) * 64;
  const int bm   = blockIdx.y * 128;
  const int bn   = blockIdx.x * 128;

  f32x8 acc[4][4] = {};

  // staging assignment: A -> one 32-elem row per thread;
  // B -> 8(k) x 4(n) register-blocked transpose per thread.
  const int arow = tid;
  const int bn0  = lane * 4;           // 4 consecutive n columns
  const int bk0  = wave * 8;           // 8 consecutive k rows

  const TA*    aptr = A + (size_t)(bm + arow) * K;
  const float* bptr = Bm + bn + bn0;

  TA     areg[32];
  float4 breg[8];

  auto load_tile = [&](int k0) {
    const TA* as = aptr + k0;
    #pragma unroll
    for (int j = 0; j < 32; ++j) areg[j] = as[j];
    #pragma unroll
    for (int kk = 0; kk < 8; ++kk)
      breg[kk] = *(const float4*)(bptr + (size_t)(k0 + bk0 + kk) * N);
  };

  load_tile(0);

  for (int k0 = 0; k0 < K; k0 += 32) {
    // ---- commit staged registers to LDS as 16B vector stores ----
    #pragma unroll
    for (int c = 0; c < 4; ++c) {
      bf16x8 v;
      #pragma unroll
      for (int j = 0; j < 8; ++j) v[j] = (__bf16)(float)areg[c * 8 + j];
      *(bf16x8*)&sA[arow][c * 8] = v;
    }
    #pragma unroll
    for (int j = 0; j < 4; ++j) {
      bf16x8 col;
      #pragma unroll
      for (int kk = 0; kk < 8; ++kk) {
        const float* bl = (const float*)&breg[kk];
        col[kk] = (__bf16)bl[j];
      }
      *(bf16x8*)&sBt[bn0 + j][bk0] = col;
    }
    __syncthreads();

    // ---- prefetch next tile into registers while WMMAs run ----
    if (k0 + 32 < K) load_tile(k0 + 32);
    if (k0 + 64 < K) {
      __builtin_prefetch(aptr + k0 + 64, 0, 1);
      __builtin_prefetch(bptr + (size_t)(k0 + 64 + bk0) * N, 0, 1);
    }

    // ---- WMMA: A-frag lane: M=l16, elems K={lh*8+e, lh*8+16+e} ----
    bf16x16 af[4];
    #pragma unroll
    for (int i = 0; i < 4; ++i) {
      const __bf16* r = &sA[wm + i * 16 + l16][0];
      af[i] = comb16(r + lh * 8, r + lh * 8 + 16);
    }
    #pragma unroll
    for (int j = 0; j < 4; ++j) {
      // B-frag lane: N=l16, elems K={lh*8+e, lh*8+16+e} (matches A-frag)
      const __bf16* r = &sBt[wn + j * 16 + l16][0];
      bf16x16 bfv = comb16(r + lh * 8, r + lh * 8 + 16);
      #pragma unroll
      for (int i = 0; i < 4; ++i) acc[i][j] = wmma_bf16(af[i], bfv, acc[i][j]);
    }
    __syncthreads();
  }

  // ---- store: C frag lane: M = r + 8*lh, N = l16 ----
  #pragma unroll
  for (int i = 0; i < 4; ++i) {
    #pragma unroll
    for (int j = 0; j < 4; ++j) {
      const int row = bm + wm + i * 16 + 8 * lh;
      const int col = bn + wn + j * 16 + l16;
      #pragma unroll
      for (int r = 0; r < 8; ++r)
        C[(size_t)(row + r) * N + col] = acc[i][j][r];
    }
  }
}

// ---------------------------------------------------------------------------
// Smoothing (2-tap temporal conv on K,V) + RoPE (Q,K) + f32 -> bf16.
// One block per (b,s) token row of the packed QKV tensor.
// ---------------------------------------------------------------------------
__global__ __launch_bounds__(256) void smooth_rope_kernel(
    const float* __restrict__ qkv, const int* __restrict__ positions,
    const float* __restrict__ conv_k, const float* __restrict__ conv_v,
    __bf16* __restrict__ Qo, __bf16* __restrict__ Ko, __bf16* __restrict__ Vo)
{
  const int bs  = blockIdx.x;
  const int s   = bs % S_;
  const int tid = threadIdx.x;
  const float pos = (float)positions[bs];
  const float* row  = qkv + (size_t)bs * F_;
  const float* prow = row - F_;            // only dereferenced when s > 0
  const bool hp = (s > 0);

  // ---- Q: RoPE only (H*64 rotation pairs) ----
  for (int i = tid; i < H_ * 64; i += 256) {
    const int hh = i >> 6, dd = i & 63;
    const float ang = pos * __powf(10000.0f, -(float)dd * (1.0f / 64.0f));
    float sn, c; __sincosf(ang, &sn, &c);
    const float* q = row + hh * D_;
    const float x1 = q[dd], x2 = q[dd + 64];
    const size_t o = (size_t)bs * (H_ * D_) + hh * D_ + dd;
    Qo[o]      = (__bf16)(x1 * c - x2 * sn);
    Qo[o + 64] = (__bf16)(x2 * c + x1 * sn);
  }
  // ---- K: smooth then RoPE (KV*64 pairs) ----
  for (int i = tid; i < KV_ * 64; i += 256) {
    const int hh = i >> 6, dd = i & 63;
    const float f0 = conv_k[hh], f1 = conv_k[KV_ + hh];
    const float* kc = row  + H_ * D_ + hh * D_;
    const float* kp = prow + H_ * D_ + hh * D_;
    const float x1 = kc[dd]      * f1 + (hp ? kp[dd]      * f0 : 0.f);
    const float x2 = kc[dd + 64] * f1 + (hp ? kp[dd + 64] * f0 : 0.f);
    const float ang = pos * __powf(10000.0f, -(float)dd * (1.0f / 64.0f));
    float sn, c; __sincosf(ang, &sn, &c);
    const size_t o = (size_t)bs * (KV_ * D_) + hh * D_ + dd;
    Ko[o]      = (__bf16)(x1 * c - x2 * sn);
    Ko[o + 64] = (__bf16)(x2 * c + x1 * sn);
  }
  // ---- V: smooth only (KV*128 elems) ----
  for (int i = tid; i < KV_ * D_; i += 256) {
    const int hh = i >> 7;
    const float f0 = conv_v[hh], f1 = conv_v[KV_ + hh];
    const float* vc = row  + (H_ + KV_) * D_;
    const float* vp = prow + (H_ + KV_) * D_;
    const float x = vc[i] * f1 + (hp ? vp[i] * f0 : 0.f);
    Vo[(size_t)bs * (KV_ * D_) + i] = (__bf16)x;
  }
}

// ---------------------------------------------------------------------------
// Flash attention with sliding window, bf16 WMMA for QK^T and PV.
// Grid: (S/64, B*H). Block = 128 threads (4 waves), each wave owns 16 q rows.
// Double-buffered pipeline: tile i+1's K DMA (global_load_async_to_lds_b128,
// ASYNCcnt) and V global loads run concurrently with tile i's WMMAs.
// ---------------------------------------------------------------------------
__global__ __launch_bounds__(128) void attn_wmma_kernel(
    const __bf16* __restrict__ Qb, const __bf16* __restrict__ Kb,
    const __bf16* __restrict__ Vb, __bf16* __restrict__ Ob)
{
  __shared__ __bf16 sK [2][32][136];   // [buf][key][d]   (+8 pad)
  __shared__ __bf16 sVt[2][128][40];   // [buf][d][key]   transposed (+8 pad)
  __shared__ __bf16 sP [4][16][32];    // per-wave P staging (C- -> A-layout)

  const int tid  = threadIdx.x;
  const int lane = tid & 31, wave = tid >> 5;
  const int l16  = lane & 15, lh = lane >> 4;
  const int bh   = blockIdx.y;
  const int b    = bh / H_, h = bh % H_;
  const int kvh  = h / (H_ / KV_);
  const int qt   = blockIdx.x;
  const int q0   = qt * 64 + wave * 16;   // this wave's query base

  // ---- Q fragments (4 K-chunks over D=128) held in registers ----
  bf16x16 qf[4];
  {
    const __bf16* qrow = Qb + ((size_t)(b * S_ + q0 + l16) * H_ + h) * D_;
    #pragma unroll
    for (int c = 0; c < 4; ++c)
      qf[c] = comb16(qrow + c * 32 + lh * 8, qrow + c * 32 + lh * 8 + 16);
  }

  f32x8 o_acc[8] = {};
  float m_r[8], l_r[8];
  #pragma unroll
  for (int r = 0; r < 8; ++r) { m_r[r] = -1e30f; l_r[r] = 0.f; }

  int klo = qt * 64 - (WIN_ - 1);
  if (klo < 0) klo = 0;
  klo &= ~31;
  const int khi  = qt * 64 + 64;
  const int nblk = (khi - klo) >> 5;
  const int kr  = tid >> 2;            // K staging: key row (0..31)
  const int dp  = (tid & 3) * 32;      // K staging: d-chunk
  const int vk0 = wave * 8;            // V staging: 8 keys
  const int vd0 = lane * 4;            // V staging: 4 d columns

  auto issueK = [&](int k0, int buf) {   // async DMA: 4 x 16B per thread
    const __bf16* ks = Kb + ((size_t)(b * S_ + k0 + kr) * KV_ + kvh) * D_ + dp;
    #pragma unroll
    for (int j = 0; j < 4; ++j)
      async_copy_b128(ks + j * 8, &sK[buf][kr][dp + j * 8]);
  };
  auto loadV = [&](int k0, bf16x4* rows) {
    #pragma unroll
    for (int kk = 0; kk < 8; ++kk)
      rows[kk] = *(const bf16x4*)(Vb +
          ((size_t)(b * S_ + k0 + vk0 + kk) * KV_ + kvh) * D_ + vd0);
  };

  // ---- pipeline prologue: tile 0 in flight ----
  bf16x4 vregs[8];
  issueK(klo, 0);
  loadV(klo, vregs);

  for (int i = 0; i < nblk; ++i) {
    const int k0  = klo + i * 32;
    const int buf = i & 1;

    // ---- commit V regs (register-blocked 8x4 transpose, 16B stores) ----
    #pragma unroll
    for (int j = 0; j < 4; ++j) {
      bf16x8 col;
      #pragma unroll
      for (int kk = 0; kk < 8; ++kk) col[kk] = vregs[kk][j];
      *(bf16x8*)&sVt[buf][vd0 + j][vk0] = col;
    }
    wait_asynccnt0();          // tile i's K DMA complete
    __syncthreads();           // staging of buf visible to all waves

    // ---- kick off tile i+1 into the other buffer (overlaps compute) ----
    if (i + 1 < nblk) { issueK(k0 + 32, buf ^ 1); loadV(k0 + 32, vregs); }

    // ---- S = Q * K^T : two 16x16 score frags (keys 0-15 / 16-31) ----
    f32x8 sacc[2] = {};
    #pragma unroll
    for (int nb = 0; nb < 2; ++nb) {
      #pragma unroll
      for (int c = 0; c < 4; ++c) {
        const __bf16* kp = &sK[buf][nb * 16 + l16][c * 32 + lh * 8];
        sacc[nb] = wmma_bf16(qf[c], comb16(kp, kp + 16), sacc[nb]);
      }
    }

    // ---- online softmax (row = r + 8*lh, col = l16 across 16 lanes) ----
    const float scale = 0.0883883476483184f;   // 1/sqrt(128)
    #pragma unroll
    for (int r = 0; r < 8; ++r) {
      const int qi  = q0 + r + 8 * lh;
      const int ki0 = k0 + l16, ki1 = k0 + 16 + l16;
      const float s0 = sacc[0][r] * scale;
      const float s1 = sacc[1][r] * scale;
      const bool ok0 = (ki0 <= qi) && (qi - ki0 < WIN_);
      const bool ok1 = (ki1 <= qi) && (qi - ki1 < WIN_);
      float v = fmaxf(ok0 ? s0 : -1e30f, ok1 ? s1 : -1e30f);
      v = fmaxf(v, __shfl_xor(v, 1, 32));
      v = fmaxf(v, __shfl_xor(v, 2, 32));
      v = fmaxf(v, __shfl_xor(v, 4, 32));
      v = fmaxf(v, __shfl_xor(v, 8, 32));      // stays within 16-lane half
      const float mnew = fmaxf(m_r[r], v);
      const float corr = __expf(m_r[r] - mnew);
      const float p0 = ok0 ? __expf(s0 - mnew) : 0.f;   // masked -> exact 0
      const float p1 = ok1 ? __expf(s1 - mnew) : 0.f;
      float ps = p0 + p1;
      ps += __shfl_xor(ps, 1, 32);
      ps += __shfl_xor(ps, 2, 32);
      ps += __shfl_xor(ps, 4, 32);
      ps += __shfl_xor(ps, 8, 32);
      l_r[r] = l_r[r] * corr + ps;
      m_r[r] = mnew;
      #pragma unroll
      for (int dj = 0; dj < 8; ++dj) o_acc[dj][r] *= corr;
      sP[wave][r + 8 * lh][l16]      = (__bf16)p0;
      sP[wave][r + 8 * lh][16 + l16] = (__bf16)p1;
    }
    __syncthreads();   // order sP C-layout stores vs A-layout reloads

    // ---- O += P(16x32) * V(32x128) ----
    bf16x16 pf = comb16(&sP[wave][l16][lh * 8], &sP[wave][l16][lh * 8 + 16]);
    #pragma unroll
    for (int dj = 0; dj < 8; ++dj) {
      const __bf16* vr = &sVt[buf][dj * 16 + l16][lh * 8];
      o_acc[dj] = wmma_bf16(pf, comb16(vr, vr + 16), o_acc[dj]);
    }
  }

  // ---- normalize + store bf16 [B,S,H,D] ----
  #pragma unroll
  for (int dj = 0; dj < 8; ++dj) {
    #pragma unroll
    for (int r = 0; r < 8; ++r) {
      const float val = o_acc[dj][r] / l_r[r];
      const size_t o =
          ((size_t)(b * S_ + q0 + r + 8 * lh) * H_ + h) * D_ + dj * 16 + l16;
      Ob[o] = (__bf16)val;
    }
  }
}

// ---------------------------------------------------------------------------
// Launcher
// ---------------------------------------------------------------------------
extern "C" void kernel_launch(void* const* d_in, const int* in_sizes, int n_in,
                              void* d_out, int out_size, void* d_ws, size_t ws_size,
                              hipStream_t stream) {
  const float* hidden    = (const float*)d_in[0];
  const int*   positions = (const int*)d_in[1];
  const float* w_pack    = (const float*)d_in[2];
  const float* w_o       = (const float*)d_in[3];
  const float* conv_k    = (const float*)d_in[4];
  const float* conv_v    = (const float*)d_in[5];
  float*       out       = (float*)d_out;

  // workspace layout (bytes)
  char* ws = (char*)d_ws;
  constexpr size_t qkv_bytes = (size_t)M_ * F_ * sizeof(float);        // 100.7 MB
  constexpr size_t q_bytes   = (size_t)M_ * H_  * D_ * sizeof(__bf16); //  33.6 MB
  constexpr size_t k_bytes   = (size_t)M_ * KV_ * D_ * sizeof(__bf16); //   8.4 MB
  constexpr size_t v_bytes   = k_bytes;
  float*  qkv = (float*)ws;
  __bf16* qb  = (__bf16*)(ws + qkv_bytes);
  __bf16* kb  = (__bf16*)(ws + qkv_bytes + q_bytes);
  __bf16* vb  = (__bf16*)(ws + qkv_bytes + q_bytes + k_bytes);
  __bf16* ab  = (__bf16*)(ws + qkv_bytes + q_bytes + k_bytes + v_bytes);
  (void)in_sizes; (void)n_in; (void)out_size; (void)ws_size;

  // 1) QKV pack GEMM (f32 in, bf16 WMMA, f32 out)
  {
    dim3 grid(F_ / 128, M_ / 128);   // 48 x 32
    gemm_bf16_wmma_kernel<float><<<grid, 128, 0, stream>>>(
        hidden, w_pack, qkv, M_, F_, HID_);
  }
  // 2) smoothing + RoPE + bf16 cast
  smooth_rope_kernel<<<M_, 256, 0, stream>>>(qkv, positions, conv_k, conv_v,
                                             qb, kb, vb);
  // 3) sliding-window flash attention (bf16 WMMA + double-buffered async DMA)
  {
    dim3 grid(S_ / 64, B_ * H_);     // 32 x 64
    attn_wmma_kernel<<<grid, 128, 0, stream>>>(qb, kb, vb, ab);
  }
  // 4) output projection GEMM (bf16 A, f32 W, f32 out)
  {
    dim3 grid(HID_ / 128, M_ / 128); // 32 x 32
    gemm_bf16_wmma_kernel<__bf16><<<grid, 128, 0, stream>>>(
        ab, w_o, out, M_, HID_, HID_);
  }
}
